// LSTMAutoEncoder_27917287424773
// MI455X (gfx1250) — matrix operand order repaired and checked by reading, not compiled, not run on hardware
//
#include <hip/hip_runtime.h>
#include <hip/hip_bf16.h>

// ---------------------------------------------------------------------------
// LSTM autoencoder for MI455X (gfx1250, wave32).
// Persistent kernel, bf16 WMMA (v_wmma_f32_16x16x32_bf16).
// - 32 blocks x 16 hidden units (block-exclusive gate slice i/f/g/o)
// - 8 waves x 16 batch rows per block (B=128)
// - weights staged in LDS once per phase (enc, then dec): B-fragments come
//   from ds_load on the serial critical path instead of L2
// - cell state c lives in registers for all 1024 steps
// - grid-wide sync via monotonic L2 atomic barrier
// ---------------------------------------------------------------------------

typedef __bf16 bf16_t;
typedef __attribute__((ext_vector_type(16))) __bf16 v16bf;
typedef __attribute__((ext_vector_type(8)))  __bf16 v8bf;
typedef __attribute__((ext_vector_type(8)))  float  v8f;

#define B_      128
#define T_      512
#define F_      64
#define H_      512
#define NBLK    32          // H_/16 hidden-slice blocks
#define THREADS 256         // 8 waves (wave32)

// LDS layout (bf16 elements)
#define S_IH0_OFF   0                       // 64 x 64   (layer0 ih slice)
#define S_HH0_OFF   (S_IH0_OFF + 64 * 64)   // 64 x 512
#define S_IH1_OFF   (S_HH0_OFF + 64 * 512)  // 64 x 512
#define S_HH1_OFF   (S_IH1_OFF + 64 * 512)  // 64 x 512
#define S_LIN_OFF   (S_HH1_OFF + 64 * 512)  // 16 x 512  (blocks 0..3)
#define S_TOTAL     (S_LIN_OFF + 16 * 512)
#define SMEM_BYTES  (S_TOTAL * 2)           // 221184 B = 216 KB (< 320 KB/WGP)

union V16U { v16bf v; v8bf h[2]; };

// ---- WMMA fragment loads ---------------------------------------------------
// A (16x32 bf16, activations, row-major, K contiguous):
//   lane L<16 : row m0+L,    K = {k0..k0+7}    and {k0+16..k0+23}
//   lane L>=16: row m0+L-16, K = {k0+8..k0+15} and {k0+24..k0+31}
__device__ __forceinline__ v16bf load_a_frag(const bf16_t* __restrict__ base,
                                             int ld, int k0, int lane, int m0) {
  const int row = m0 + (lane & 15);
  const int kk  = k0 + ((lane >> 4) << 3);
  const bf16_t* p = base + (size_t)row * (size_t)ld + kk;
  V16U u;
  u.h[0] = *reinterpret_cast<const v8bf*>(p);        // 16B
  u.h[1] = *reinterpret_cast<const v8bf*>(p + 16);   // 16B
  return u.v;
}

// B (32x16 bf16) from an LDS-staged [64 x ldw] slice (rows = 4 gate tiles):
//   lane n<16 : col n,    K = {k0..k0+7}    and {k0+16..k0+23}
//   lane n>=16: col n-16, K = {k0+8..k0+15} and {k0+24..k0+31}
__device__ __forceinline__ v16bf load_b_lds(const bf16_t* __restrict__ S,
                                            int ldw, int gtile, int k0, int lane) {
  const int r  = (gtile << 4) + (lane & 15);
  const int kk = k0 + ((lane >> 4) << 3);
  V16U u;
  u.h[0] = *reinterpret_cast<const v8bf*>(S + r * ldw + kk);
  u.h[1] = *reinterpret_cast<const v8bf*>(S + r * ldw + kk + 16);
  return u.v;
}

__device__ __forceinline__ v8f wmma_bf16(v16bf a, v16bf b, v8f c) {
  return __builtin_amdgcn_wmma_f32_16x16x32_bf16(false, a, false, b,
                                                 (short)0, c, false, false);
}

__device__ __forceinline__ float sigmoidf_(float x) {
  return 1.0f / (1.0f + __expf(-x));
}

// ---- grid-wide barrier (monotonic arrive + generation, agent scope) --------
__device__ __forceinline__ void grid_barrier(unsigned* __restrict__ bar,
                                             unsigned nblk, unsigned& phase) {
  __syncthreads();
  if (threadIdx.x == 0) {
    ++phase;
    __threadfence();
    unsigned arrived = __hip_atomic_fetch_add(&bar[0], 1u, __ATOMIC_ACQ_REL,
                                              __HIP_MEMORY_SCOPE_AGENT) + 1u;
    if (arrived == phase * nblk) {
      __hip_atomic_fetch_add(&bar[1], 1u, __ATOMIC_RELEASE,
                             __HIP_MEMORY_SCOPE_AGENT);
    } else {
      while (__hip_atomic_load(&bar[1], __ATOMIC_ACQUIRE,
                               __HIP_MEMORY_SCOPE_AGENT) < phase) {
        __builtin_amdgcn_s_sleep(2);
      }
    }
    __threadfence();
  }
  __syncthreads();
}

// ---- stage a [64 x ldw] gate slice of W[4H x ldw] into LDS -----------------
__device__ __forceinline__ void stage_w(const bf16_t* __restrict__ W, int ldw,
                                        int hb, bf16_t* __restrict__ dst, int tid) {
  const int total = 64 * ldw;
  for (int i = tid * 8; i < total; i += THREADS * 8) {
    const int r = i / ldw;
    const int k = i - r * ldw;
    const int grow = (r >> 4) * H_ + hb + (r & 15);   // gate-major rows
    *reinterpret_cast<v8bf*>(dst + i) =
        *reinterpret_cast<const v8bf*>(W + (size_t)grow * ldw + k);
  }
}

__device__ __forceinline__ void stage_lin(const bf16_t* __restrict__ W, int nq,
                                          bf16_t* __restrict__ dst, int tid) {
  const int total = 16 * H_;
  for (int i = tid * 8; i < total; i += THREADS * 8) {
    const int r = i / H_;
    const int k = i - r * H_;
    *reinterpret_cast<v8bf*>(dst + i) =
        *reinterpret_cast<const v8bf*>(W + (size_t)(nq + r) * H_ + k);
  }
}

// ---- one LSTM cell, this block's 16-hidden-unit slice ----------------------
__device__ __forceinline__ void lstm_slice(
    const bf16_t* __restrict__ Xin, int ldx, int Kx,
    const bf16_t* __restrict__ Hin,           // [B,H] previous h of this layer
    const bf16_t* __restrict__ sWih,          // LDS [64 x Kx]
    const bf16_t* __restrict__ sWhh,          // LDS [64 x H]
    float bi, float bf, float bg, float bo,   // per-lane bias scalars
    float (&creg)[8],                         // register-resident cell state
    bf16_t*       __restrict__ Hout,          // [B,H] bf16 new h
    int col, int rbase, int lane, int m0)
{
  v8f acc[4] = {};
  for (int k0 = 0; k0 < Kx; k0 += 32) {
    v16bf a = load_a_frag(Xin, ldx, k0, lane, m0);
#pragma unroll
    for (int g = 0; g < 4; ++g)
      acc[g] = wmma_bf16(a, load_b_lds(sWih, Kx, g, k0, lane), acc[g]);
  }
  for (int k0 = 0; k0 < H_; k0 += 32) {
    v16bf a = load_a_frag(Hin, H_, k0, lane, m0);
#pragma unroll
    for (int g = 0; g < 4; ++g)
      acc[g] = wmma_bf16(a, load_b_lds(sWhh, H_, g, k0, lane), acc[g]);
  }
  // C/D layout: lane n<16 -> col n, rows m0+r; lane n>=16 -> col n-16, rows m0+8+r
#pragma unroll
  for (int r = 0; r < 8; ++r) {
    const float iv = sigmoidf_(acc[0][r] + bi);
    const float fv = sigmoidf_(acc[1][r] + bf);
    const float gv = tanhf(acc[2][r] + bg);
    const float ov = sigmoidf_(acc[3][r] + bo);
    const float cn = fv * creg[r] + iv * gv;
    creg[r] = cn;
    Hout[(rbase + r) * H_ + col] = (bf16_t)(ov * tanhf(cn));
  }
}

// ---- decoder output projection: out = h_top @ lin_W.T + lin_b --------------
__device__ __forceinline__ void proj_slice(
    const bf16_t* __restrict__ Htop,   // [B,H]
    const bf16_t* __restrict__ sLin,   // LDS [16 x H]
    float bv,                          // per-lane lin bias
    float*        __restrict__ outp,   // d_out + tout*F ; row stride T*F
    bf16_t*       __restrict__ dec_in, // [B,F] bf16 feedback
    int nq, int rbase, int lane, int m0)
{
  v8f acc = {};
  for (int k0 = 0; k0 < H_; k0 += 32) {
    v16bf a = load_a_frag(Htop, H_, k0, lane, m0);
    acc = wmma_bf16(a, load_b_lds(sLin, H_, 0, k0, lane), acc);
  }
  const int col = nq + (lane & 15);
#pragma unroll
  for (int r = 0; r < 8; ++r) {
    const int row = rbase + r;
    const float v = acc[r] + bv;
    outp[(size_t)row * (size_t)(T_ * F_) + col] = v;
    dec_in[row * F_ + col] = (bf16_t)v;
  }
}

// ---- kernel params ---------------------------------------------------------
struct KParams {
  const bf16_t *xb;                               // [B,T,F] bf16
  const bf16_t *eW0, *eU0, *eW1, *eU1;            // encoder weights bf16
  const bf16_t *dW0, *dU0, *dW1, *dU1;            // decoder weights bf16
  const bf16_t *lW;                               // [F,H] bf16
  const float  *eb0, *eb1, *db0, *db1, *lb;       // fp32 biases
  const float  *c0in;                             // [2,B,H] fp32 initial c
  bf16_t *h;         // [2 layer][2 parity][B,H] bf16
  bf16_t *dec_in;    // [B,F] bf16 autoregressive feedback
  float  *out;       // [B,T,F] fp32
  unsigned *bar;     // barrier: [0]=arrive, [1]=generation
};

__global__ void __launch_bounds__(THREADS, 1)
lstm_autoenc_persistent(KParams p)
{
  extern __shared__ bf16_t smem[];
  bf16_t* sIh  = smem + S_IH0_OFF;
  bf16_t* sHh0 = smem + S_HH0_OFF;
  bf16_t* sIh1 = smem + S_IH1_OFF;
  bf16_t* sHh1 = smem + S_HH1_OFF;
  bf16_t* sLin = smem + S_LIN_OFF;

  const int tid  = threadIdx.x;
  const int lane = tid & 31;
  const int m0   = (tid >> 5) << 4;           // wave -> 16 batch rows
  const int hb   = blockIdx.x << 4;           // block -> 16 hidden units
  const int col  = hb + (lane & 15);
  const int rbase = m0 + ((lane >> 4) << 3);
  const size_t BH = (size_t)B_ * H_;
  unsigned phase = 0;

  // per-lane bias scalars, loaded once
  const float ei = p.eb0[0 * H_ + col], ef = p.eb0[1 * H_ + col],
              eg = p.eb0[2 * H_ + col], eo = p.eb0[3 * H_ + col];
  const float Ei = p.eb1[0 * H_ + col], Ef = p.eb1[1 * H_ + col],
              Eg = p.eb1[2 * H_ + col], Eo = p.eb1[3 * H_ + col];
  const float di = p.db0[0 * H_ + col], df = p.db0[1 * H_ + col],
              dg = p.db0[2 * H_ + col], dd = p.db0[3 * H_ + col];
  const float Di = p.db1[0 * H_ + col], Df = p.db1[1 * H_ + col],
              Dg = p.db1[2 * H_ + col], Do = p.db1[3 * H_ + col];
  const int   nq  = (blockIdx.x < 4) ? (blockIdx.x << 4) : 0;
  const float lbv = (blockIdx.x < 4) ? p.lb[nq + (lane & 15)] : 0.0f;

  // cell state in registers for the entire run (block/lane-exclusive)
  float c0r[8], c1r[8];
#pragma unroll
  for (int r = 0; r < 8; ++r) {
    c0r[r] = p.c0in[0 * BH + (size_t)(rbase + r) * H_ + col];
    c1r[r] = p.c0in[1 * BH + (size_t)(rbase + r) * H_ + col];
  }

  // stage encoder weight slices into LDS (once)
  stage_w(p.eW0, F_, hb, sIh,  tid);
  stage_w(p.eU0, H_, hb, sHh0, tid);
  stage_w(p.eW1, H_, hb, sIh1, tid);
  stage_w(p.eU1, H_, hb, sHh1, tid);
  __syncthreads();

  int pp = 0;   // parity: read h[pp], write h[pp^1]

  // -------- encoder: consume x, keep final (h, c) --------
  for (int t = 0; t < T_; ++t) {
    const bf16_t* h0r = p.h + (size_t)pp * BH;
    bf16_t*       h0w = p.h + (size_t)(pp ^ 1) * BH;
    const bf16_t* h1r = p.h + (size_t)(2 + pp) * BH;
    bf16_t*       h1w = p.h + (size_t)(2 + (pp ^ 1)) * BH;

    if (t + 1 < T_)   // warm next timestep's x rows into cache
      __builtin_prefetch(p.xb + (size_t)(m0 + (lane & 15)) * (T_ * F_) +
                             (size_t)(t + 1) * F_, 0, 1);

    lstm_slice(p.xb + (size_t)t * F_, T_ * F_, F_, h0r, sIh, sHh0,
               ei, ef, eg, eo, c0r, h0w, col, rbase, lane, m0);
    grid_barrier(p.bar, NBLK, phase);
    lstm_slice(h0w, H_, H_, h1r, sIh1, sHh1,
               Ei, Ef, Eg, Eo, c1r, h1w, col, rbase, lane, m0);
    grid_barrier(p.bar, NBLK, phase);
    pp ^= 1;
  }

  // re-stage LDS with decoder weight slices
  stage_w(p.dW0, F_, hb, sIh,  tid);
  stage_w(p.dU0, H_, hb, sHh0, tid);
  stage_w(p.dW1, H_, hb, sIh1, tid);
  stage_w(p.dU1, H_, hb, sHh1, tid);
  if (blockIdx.x < 4) stage_lin(p.lW, nq, sLin, tid);
  __syncthreads();

  // -------- decoder: autoregressive, outputs in reverse time order --------
  for (int t = 0; t < T_; ++t) {
    const bf16_t* h0r = p.h + (size_t)pp * BH;
    bf16_t*       h0w = p.h + (size_t)(pp ^ 1) * BH;
    const bf16_t* h1r = p.h + (size_t)(2 + pp) * BH;
    bf16_t*       h1w = p.h + (size_t)(2 + (pp ^ 1)) * BH;

    lstm_slice(p.dec_in, F_, F_, h0r, sIh, sHh0,
               di, df, dg, dd, c0r, h0w, col, rbase, lane, m0);
    grid_barrier(p.bar, NBLK, phase);
    lstm_slice(h0w, H_, H_, h1r, sIh1, sHh1,
               Di, Df, Dg, Do, c1r, h1w, col, rbase, lane, m0);
    grid_barrier(p.bar, NBLK, phase);
    if (blockIdx.x < 4) {   // F=64 -> 4 col tiles
      proj_slice(h1w, sLin, lbv, p.out + (size_t)(T_ - 1 - t) * F_,
                 p.dec_in, nq, rbase, lane, m0);
    }
    grid_barrier(p.bar, NBLK, phase);   // feedback visible before next step
    pp ^= 1;
  }
}

// ---- prep kernels ----------------------------------------------------------
__global__ void cvt_f32_bf16(const float* __restrict__ s,
                             bf16_t* __restrict__ d, int n) {
  int i = blockIdx.x * blockDim.x + threadIdx.x;
  if (i < n) d[i] = (bf16_t)s[i];
}

__global__ void init_state(const float* __restrict__ h0,
                           const float* __restrict__ x,
                           bf16_t* __restrict__ h,
                           bf16_t* __restrict__ dec_in,
                           unsigned* __restrict__ bar) {
  int i = blockIdx.x * blockDim.x + threadIdx.x;
  const int n = 2 * B_ * H_;
  if (i < n) {
    int l = i / (B_ * H_);
    int j = i - l * (B_ * H_);
    h[(l * 2 + 0) * (B_ * H_) + j] = (bf16_t)h0[i];   // parity 0
  }
  if (i < B_ * F_) {
    int b = i / F_, f = i - b * F_;
    dec_in[i] = (bf16_t)x[(size_t)b * T_ * F_ + (size_t)(T_ - 1) * F_ + f];
  }
  if (i < 4) bar[i] = 0u;
}

// ---------------------------------------------------------------------------
extern "C" void kernel_launch(void* const* d_in, const int* in_sizes, int n_in,
                              void* d_out, int out_size, void* d_ws, size_t ws_size,
                              hipStream_t stream) {
  (void)in_sizes; (void)n_in; (void)out_size; (void)ws_size;

  const float* x    = (const float*)d_in[0];
  const float* h0   = (const float*)d_in[1];
  const float* c0   = (const float*)d_in[2];
  const float* eW0f = (const float*)d_in[3];
  const float* eU0f = (const float*)d_in[4];
  const float* eb0  = (const float*)d_in[5];
  const float* eW1f = (const float*)d_in[6];
  const float* eU1f = (const float*)d_in[7];
  const float* eb1  = (const float*)d_in[8];
  const float* dW0f = (const float*)d_in[9];
  const float* dU0f = (const float*)d_in[10];
  const float* db0  = (const float*)d_in[11];
  const float* dW1f = (const float*)d_in[12];
  const float* dU1f = (const float*)d_in[13];
  const float* db1  = (const float*)d_in[14];
  const float* lWf  = (const float*)d_in[15];
  const float* lb   = (const float*)d_in[16];

  // workspace carve-up (256B aligned)
  char* ws = (char*)d_ws;
  size_t off = 0;
  auto take = [&](size_t bytes) -> void* {
    void* r = ws + off;
    off = (off + bytes + 255) & ~(size_t)255;
    return r;
  };
  bf16_t* xb  = (bf16_t*)take((size_t)B_ * T_ * F_ * 2);
  bf16_t* eW0 = (bf16_t*)take((size_t)4 * H_ * F_ * 2);
  bf16_t* eU0 = (bf16_t*)take((size_t)4 * H_ * H_ * 2);
  bf16_t* eW1 = (bf16_t*)take((size_t)4 * H_ * H_ * 2);
  bf16_t* eU1 = (bf16_t*)take((size_t)4 * H_ * H_ * 2);
  bf16_t* dW0 = (bf16_t*)take((size_t)4 * H_ * F_ * 2);
  bf16_t* dU0 = (bf16_t*)take((size_t)4 * H_ * H_ * 2);
  bf16_t* dW1 = (bf16_t*)take((size_t)4 * H_ * H_ * 2);
  bf16_t* dU1 = (bf16_t*)take((size_t)4 * H_ * H_ * 2);
  bf16_t* lW  = (bf16_t*)take((size_t)F_ * H_ * 2);
  bf16_t* hbuf  = (bf16_t*)take((size_t)4 * B_ * H_ * 2);  // 2 layers x 2 parity
  bf16_t* decin = (bf16_t*)take((size_t)B_ * F_ * 2);
  unsigned* bar = (unsigned*)take(4 * sizeof(unsigned));

  auto cvt = [&](const float* s, bf16_t* d, int n) {
    cvt_f32_bf16<<<(n + 255) / 256, 256, 0, stream>>>(s, d, n);
  };
  cvt(x,    xb,  B_ * T_ * F_);
  cvt(eW0f, eW0, 4 * H_ * F_);
  cvt(eU0f, eU0, 4 * H_ * H_);
  cvt(eW1f, eW1, 4 * H_ * H_);
  cvt(eU1f, eU1, 4 * H_ * H_);
  cvt(dW0f, dW0, 4 * H_ * F_);
  cvt(dU0f, dU0, 4 * H_ * H_);
  cvt(dW1f, dW1, 4 * H_ * H_);
  cvt(dU1f, dU1, 4 * H_ * H_);
  cvt(lWf,  lW,  F_ * H_);

  {
    const int n = 2 * B_ * H_;
    init_state<<<(n + 255) / 256, 256, 0, stream>>>(h0, x, hbuf, decin, bar);
  }

  KParams p;
  p.xb = xb;
  p.eW0 = eW0; p.eU0 = eU0; p.eW1 = eW1; p.eU1 = eU1;
  p.dW0 = dW0; p.dU0 = dU0; p.dW1 = dW1; p.dU1 = dU1;
  p.lW = lW;
  p.eb0 = eb0; p.eb1 = eb1; p.db0 = db0; p.db1 = db1; p.lb = lb;
  p.c0in = c0;
  p.h = hbuf; p.dec_in = decin;
  p.out = (float*)d_out;
  p.bar = bar;

  // allow 216KB dynamic LDS (gfx1250: up to 320KB per workgroup)
  hipFuncSetAttribute((const void*)lstm_autoenc_persistent,
                      hipFuncAttributeMaxDynamicSharedMemorySize, SMEM_BYTES);

  lstm_autoenc_persistent<<<NBLK, THREADS, SMEM_BYTES, stream>>>(p);
}
